// GRU_47949014892908
// MI455X (gfx1250) — compile-verified, matrix operand order checked by reading, not run on hardware
//
#include <hip/hip_runtime.h>
#include <hip/hip_bf16.h>

#define TT 128
#define BB 32
#define EE 512
#define HH 1024
#define VV 32000
#define LL 2

typedef __bf16 bf16_t;
typedef __attribute__((ext_vector_type(16))) __bf16 v16bf;
typedef __attribute__((ext_vector_type(8)))  __bf16 v8bf;
typedef __attribute__((ext_vector_type(8)))  float  v8f;

// ---------- helpers ----------

static __device__ __forceinline__ bf16_t f2bf(float f) {
    union { float f; unsigned u; } v; v.f = f;
    unsigned r = v.u + 0x7FFFu + ((v.u >> 16) & 1u);   // round-to-nearest-even
    unsigned short h = (unsigned short)(r >> 16);
    bf16_t out;
    __builtin_memcpy(&out, &h, sizeof(out));
    return out;
}

static __device__ __forceinline__ float sigmoidf_(float x) {
    return 1.0f / (1.0f + __expf(-x));
}

// CDNA5 wave32 16-bit A/B operand tile loader:
//   row = row0 + (lane & 15); lanes >= 16 take the K = +8 / +24 halves.
// Two 16-byte vector loads per lane (global_load_b128 or ds_load_b128).
static __device__ __forceinline__ v16bf load_tile(const bf16_t* __restrict__ p,
                                                  int row0, int k0, int ld) {
    int lane = threadIdx.x & 31;
    int r  = row0 + (lane & 15);
    int kb = k0 + ((lane >> 4) << 3);
    const bf16_t* q = p + (size_t)r * (size_t)ld + kb;
    v8bf lo = *(const v8bf*)q;
    v8bf hi = *(const v8bf*)(q + 16);
    v16bf o;
#pragma unroll
    for (int i = 0; i < 8; ++i) { o[i] = lo[i]; o[i + 8] = hi[i]; }
    return o;
}

static __device__ __forceinline__ v8f wmma_bf16(v16bf a, v16bf b, v8f c) {
    return __builtin_amdgcn_wmma_f32_16x16x32_bf16(
        /*neg_a=*/false, a, /*neg_b=*/false, b,
        /*c_mod=*/(short)0, c, /*reuse_a=*/false, /*reuse_b=*/false);
}

// Async global->LDS 16B copy (gfx1250 GVS form; tracked with ASYNCcnt).
static __device__ __forceinline__ void async_b128(unsigned lds_off,
                                                  unsigned gbl_off,
                                                  unsigned long long saddr) {
    asm volatile("global_load_async_to_lds_b128 %0, %1, %2"
                 :: "v"(lds_off), "v"(gbl_off), "s"(saddr)
                 : "memory");
}

static __device__ __forceinline__ void wait_async0() {
    asm volatile("s_wait_asynccnt 0x0" ::: "memory");
}

// ---------- elementwise kernels ----------

__global__ void cvt_bf16_kernel(const float* __restrict__ src,
                                bf16_t* __restrict__ dst, long n) {
    long i = (long)blockIdx.x * blockDim.x + threadIdx.x;
    if (i < n) dst[i] = f2bf(src[i]);
}

__global__ void embed_kernel(const int* __restrict__ tok,
                             const float* __restrict__ emb,
                             bf16_t* __restrict__ x) {
    long i = (long)blockIdx.x * blockDim.x + threadIdx.x;   // over T*B*E
    long row = i / EE, col = i % EE;
    int t = tok[row];
    x[i] = f2bf(emb[(size_t)t * EE + col]);
}

// ---------- LDS-staged WMMA GEMM: C[M,N] = A[M,K] @ W[N,K]^T (+bias) -------
// block = 256 threads = 8 waves (2 M-strips x 4 N-strips).
// block tile = 64 x 128, staged in LDS 64 K at a time via async-to-LDS.
// wave tile = 32 x 32 (4 accumulators). Assumes M%64==0, N%128==0, K%64==0.

__global__ __launch_bounds__(256)
void gemm_lds_kernel(const bf16_t* __restrict__ A, const bf16_t* __restrict__ W,
                     const float* __restrict__ bias, float* __restrict__ C,
                     int M, int N, int K) {
    __shared__ bf16_t As[64 * 64];     // [64 rows][64 K]
    __shared__ bf16_t Ws[128 * 64];    // [128 rows][64 K]
    (void)M;

    const int tid  = threadIdx.x;
    const int wave = tid >> 5;
    const int lane = tid & 31;
    const int wm = wave & 1;           // 32-row strip within block tile
    const int wn = wave >> 1;          // 32-col strip within block tile
    const int blockM = blockIdx.y * 64;
    const int blockN = blockIdx.x * 128;

    const unsigned lds_a = (unsigned)(size_t)(&As[0]);
    const unsigned lds_w = (unsigned)(size_t)(&Ws[0]);
    const unsigned long long a_base = (unsigned long long)(size_t)A;
    const unsigned long long w_base = (unsigned long long)(size_t)W;

    v8f acc[4] = {};
    for (int k0 = 0; k0 < K; k0 += 64) {
        // ---- stage A tile: 64 rows x 64 K = 512 x 16B chunks, 2 per thread
#pragma unroll
        for (int i = 0; i < 2; ++i) {
            int c = tid + i * 256;
            int row = c >> 3, seg = c & 7;
            unsigned goff = (unsigned)(((size_t)(blockM + row) * K + k0 + seg * 8) * 2);
            unsigned loff = lds_a + (unsigned)((row * 64 + seg * 8) * 2);
            async_b128(loff, goff, a_base);
        }
        // ---- stage W tile: 128 rows x 64 K = 1024 x 16B chunks, 4 per thread
#pragma unroll
        for (int i = 0; i < 4; ++i) {
            int c = tid + i * 256;
            int row = c >> 3, seg = c & 7;
            unsigned goff = (unsigned)(((size_t)(blockN + row) * K + k0 + seg * 8) * 2);
            unsigned loff = lds_w + (unsigned)((row * 64 + seg * 8) * 2);
            async_b128(loff, goff, w_base);
        }
        wait_async0();
        __syncthreads();

        // ---- compute from LDS: 8 WMMAs per 64-K step per wave
#pragma unroll
        for (int kk = 0; kk < 64; kk += 32) {
            v16bf a0 = load_tile(As, wm * 32,      kk, 64);
            v16bf a1 = load_tile(As, wm * 32 + 16, kk, 64);
            v16bf b0 = load_tile(Ws, wn * 32,      kk, 64);
            v16bf b1 = load_tile(Ws, wn * 32 + 16, kk, 64);
            acc[0] = wmma_bf16(a0, b0, acc[0]);
            acc[1] = wmma_bf16(a0, b1, acc[1]);
            acc[2] = wmma_bf16(a1, b0, acc[2]);
            acc[3] = wmma_bf16(a1, b1, acc[3]);
        }
        __syncthreads();   // before next stage overwrites the tiles
    }

    int rbase = blockM + wm * 32 + ((lane >> 4) << 3);
    int cbase = blockN + wn * 32 + (lane & 15);
#pragma unroll
    for (int mi = 0; mi < 2; ++mi) {
#pragma unroll
        for (int ni = 0; ni < 2; ++ni) {
            int n = cbase + ni * 16;
            float bv = bias ? bias[n] : 0.0f;
            const v8f& a = acc[mi * 2 + ni];
#pragma unroll
            for (int i = 0; i < 8; ++i)
                C[(size_t)(rbase + mi * 16 + i) * N + n] = a[i] + bv;
        }
    }
}

// ---------- GRU cell, phase 1: r & z gates + (r*h) ----------
// grid = 32 blocks (H/32), block = 128 threads = 4 waves, wave tile 16x16.
// gates = sigmoid([xproj or x@W^T] + h@U^T + b); stores z (f32) and bf16(r*h).

__global__ __launch_bounds__(128)
void gru_rz_kernel(const bf16_t* __restrict__ h_bf, const float* __restrict__ h_f,
                   const float* __restrict__ xpr, const float* __restrict__ xpz,
                   const bf16_t* __restrict__ x_bf, int xK,
                   const bf16_t* __restrict__ Wr, const bf16_t* __restrict__ Wz,
                   const bf16_t* __restrict__ Ur, const bf16_t* __restrict__ Uz,
                   const float* __restrict__ br, const float* __restrict__ bz,
                   float* __restrict__ z_out, bf16_t* __restrict__ rh_out) {
    int wave = threadIdx.x >> 5;
    int lane = threadIdx.x & 31;
    int m0 = (wave & 1) * 16;
    int n0 = blockIdx.x * 32 + (wave >> 1) * 16;
    v8f ar = {}, az = {};
    for (int k = 0; k < HH; k += 32) {
        v16bf a = load_tile(h_bf, m0, k, HH);
        ar = wmma_bf16(a, load_tile(Ur, n0, k, HH), ar);
        az = wmma_bf16(a, load_tile(Uz, n0, k, HH), az);
    }
    if (x_bf) {
        for (int k = 0; k < xK; k += 32) {
            v16bf a = load_tile(x_bf, m0, k, xK);
            ar = wmma_bf16(a, load_tile(Wr, n0, k, xK), ar);
            az = wmma_bf16(a, load_tile(Wz, n0, k, xK), az);
        }
    }
    int rbase = m0 + ((lane >> 4) << 3);
    int n = n0 + (lane & 15);
    float brv = br[n], bzv = bz[n];
#pragma unroll
    for (int i = 0; i < 8; ++i) {
        size_t idx = (size_t)(rbase + i) * HH + n;
        float pr = ar[i] + brv + (xpr ? xpr[idx] : 0.0f);
        float pz = az[i] + bzv + (xpz ? xpz[idx] : 0.0f);
        float r = sigmoidf_(pr);
        z_out[idx] = sigmoidf_(pz);
        rh_out[idx] = f2bf(r * h_f[idx]);
    }
}

// ---------- GRU cell, phase 2: h~ and state update ----------
// h~ = tanh([xproj or x@Wh^T] + (r*h)@Uh^T + bh); h' = (1-z)h + z h~

__global__ __launch_bounds__(128)
void gru_h_kernel(const bf16_t* __restrict__ rh_bf,
                  const float* __restrict__ xph,
                  const bf16_t* __restrict__ x_bf, int xK,
                  const bf16_t* __restrict__ Wh,
                  const bf16_t* __restrict__ Uh,
                  const float* __restrict__ bh,
                  const float* __restrict__ z_buf,
                  float* __restrict__ h_f,          // in/out state [B,H]
                  bf16_t* __restrict__ h_bf_out,    // bf16 state copy
                  bf16_t* __restrict__ y_out) {     // optional extra bf16 copy
    int wave = threadIdx.x >> 5;
    int lane = threadIdx.x & 31;
    int m0 = (wave & 1) * 16;
    int n0 = blockIdx.x * 32 + (wave >> 1) * 16;
    v8f ah = {};
    for (int k = 0; k < HH; k += 32) {
        v16bf a = load_tile(rh_bf, m0, k, HH);
        ah = wmma_bf16(a, load_tile(Uh, n0, k, HH), ah);
    }
    if (x_bf) {
        for (int k = 0; k < xK; k += 32) {
            v16bf a = load_tile(x_bf, m0, k, xK);
            ah = wmma_bf16(a, load_tile(Wh, n0, k, xK), ah);
        }
    }
    int rbase = m0 + ((lane >> 4) << 3);
    int n = n0 + (lane & 15);
    float bhv = bh[n];
#pragma unroll
    for (int i = 0; i < 8; ++i) {
        size_t idx = (size_t)(rbase + i) * HH + n;
        float pre = ah[i] + bhv + (xph ? xph[idx] : 0.0f);
        float ht = tanhf(pre);
        float z = z_buf[idx];
        float hn = (1.0f - z) * h_f[idx] + z * ht;
        h_f[idx] = hn;
        bf16_t hb = f2bf(hn);
        h_bf_out[idx] = hb;
        if (y_out) y_out[idx] = hb;
    }
}

// ---------- host orchestration ----------

extern "C" void kernel_launch(void* const* d_in, const int* in_sizes, int n_in,
                              void* d_out, int out_size, void* d_ws, size_t ws_size,
                              hipStream_t stream) {
    const int*   tok   = (const int*)  d_in[0];
    const float* hid0  = (const float*)d_in[1];
    const float* emb   = (const float*)d_in[2];
    const float* wr0   = (const float*)d_in[3];
    const float* wz0   = (const float*)d_in[4];
    const float* wh0   = (const float*)d_in[5];
    const float* wr1   = (const float*)d_in[6];
    const float* wz1   = (const float*)d_in[7];
    const float* wh1   = (const float*)d_in[8];
    const float* urw   = (const float*)d_in[9];
    const float* urb   = (const float*)d_in[10];
    const float* uzw   = (const float*)d_in[11];
    const float* uzb   = (const float*)d_in[12];
    const float* uhw   = (const float*)d_in[13];
    const float* uhb   = (const float*)d_in[14];
    const float* wyw   = (const float*)d_in[15];
    const float* wyb   = (const float*)d_in[16];

    char* ws = (char*)d_ws;
    size_t off = 0;
    auto alloc = [&](size_t bytes) -> void* {
        void* p = ws + off;
        off = (off + bytes + 255) & ~(size_t)255;
        return p;
    };

    bf16_t* wr0b = (bf16_t*)alloc((size_t)HH * EE * 2);
    bf16_t* wz0b = (bf16_t*)alloc((size_t)HH * EE * 2);
    bf16_t* wh0b = (bf16_t*)alloc((size_t)HH * EE * 2);
    bf16_t* wr1b = (bf16_t*)alloc((size_t)HH * HH * 2);
    bf16_t* wz1b = (bf16_t*)alloc((size_t)HH * HH * 2);
    bf16_t* wh1b = (bf16_t*)alloc((size_t)HH * HH * 2);
    bf16_t* urwb = (bf16_t*)alloc((size_t)LL * HH * HH * 2);
    bf16_t* uzwb = (bf16_t*)alloc((size_t)LL * HH * HH * 2);
    bf16_t* uhwb = (bf16_t*)alloc((size_t)LL * HH * HH * 2);
    bf16_t* wywb = (bf16_t*)alloc((size_t)VV * HH * 2);
    bf16_t* xemb = (bf16_t*)alloc((size_t)TT * BB * EE * 2);
    float*  xpR  = (float*) alloc((size_t)TT * BB * HH * 4);
    float*  xpZ  = (float*) alloc((size_t)TT * BB * HH * 4);
    float*  xpH  = (float*) alloc((size_t)TT * BB * HH * 4);
    float*  hstate = (float*)alloc((size_t)LL * BB * HH * 4);   // [L,B,H]
    bf16_t* h0b  = (bf16_t*)alloc((size_t)BB * HH * 2);
    bf16_t* h1b  = (bf16_t*)alloc((size_t)BB * HH * 2);
    bf16_t* ybf  = (bf16_t*)alloc((size_t)TT * BB * HH * 2);    // layer-1 outputs
    float*  zbuf = (float*) alloc((size_t)BB * HH * 4);
    bf16_t* rhb  = (bf16_t*)alloc((size_t)BB * HH * 2);
    (void)ws_size; (void)in_sizes; (void)n_in; (void)out_size;

    auto cvt = [&](const float* s, bf16_t* d, long n) {
        cvt_bf16_kernel<<<dim3((unsigned)((n + 255) / 256)), dim3(256), 0, stream>>>(s, d, n);
    };

    // 1. weight down-conversion (L2-resident thereafter)
    cvt(wr0, wr0b, (long)HH * EE);
    cvt(wz0, wz0b, (long)HH * EE);
    cvt(wh0, wh0b, (long)HH * EE);
    cvt(wr1, wr1b, (long)HH * HH);
    cvt(wz1, wz1b, (long)HH * HH);
    cvt(wh1, wh1b, (long)HH * HH);
    cvt(urw, urwb, (long)LL * HH * HH);
    cvt(uzw, uzwb, (long)LL * HH * HH);
    cvt(uhw, uhwb, (long)LL * HH * HH);
    cvt(wyw, wywb, (long)VV * HH);
    // initial hidden state (f32 + bf16 copies)
    cvt(hid0,            h0b, (long)BB * HH);
    cvt(hid0 + BB * HH,  h1b, (long)BB * HH);
    hipMemcpyAsync(hstate, hid0, (size_t)LL * BB * HH * 4,
                   hipMemcpyDeviceToDevice, stream);

    // 2. embedding gather -> bf16 [T*B, E]
    embed_kernel<<<dim3((unsigned)((long)TT * BB * EE / 256)), dim3(256), 0, stream>>>(
        tok, emb, xemb);

    // 3. layer-0 input projections for all timesteps at once (M=4096,N=1024,K=512)
    {
        dim3 grid(HH / 128, (TT * BB) / 64);
        gemm_lds_kernel<<<grid, 256, 0, stream>>>(xemb, wr0b, nullptr, xpR, TT * BB, HH, EE);
        gemm_lds_kernel<<<grid, 256, 0, stream>>>(xemb, wz0b, nullptr, xpZ, TT * BB, HH, EE);
        gemm_lds_kernel<<<grid, 256, 0, stream>>>(xemb, wh0b, nullptr, xpH, TT * BB, HH, EE);
    }

    // 4. sequential recurrence
    float*  hf0 = hstate;
    float*  hf1 = hstate + (size_t)BB * HH;
    const size_t uoff = (size_t)HH * HH;
    for (int t = 0; t < TT; ++t) {
        const size_t xo = (size_t)t * BB * HH;
        // ---- layer 0 (input part precomputed) ----
        gru_rz_kernel<<<32, 128, 0, stream>>>(
            h0b, hf0, xpR + xo, xpZ + xo,
            nullptr, 0, nullptr, nullptr,
            urwb, uzwb, urb, uzb, zbuf, rhb);
        gru_h_kernel<<<32, 128, 0, stream>>>(
            rhb, xpH + xo, nullptr, 0, nullptr,
            uhwb, uhb, zbuf, hf0, h0b, nullptr);
        // ---- layer 1 (input = layer-0 output, folded into the WMMA loops) ----
        gru_rz_kernel<<<32, 128, 0, stream>>>(
            h1b, hf1, nullptr, nullptr,
            h0b, HH, wr1b, wz1b,
            urwb + uoff, uzwb + uoff, urb + HH, uzb + HH, zbuf, rhb);
        gru_h_kernel<<<32, 128, 0, stream>>>(
            rhb, nullptr, h0b, HH, wh1b,
            uhwb + uoff, uhb + HH, zbuf, hf1, h1b, ybf + xo);
    }

    // 5. logits GEMM: [T*B, H] x [V, H]^T + bias -> d_out (M=4096, N=32000, K=1024)
    {
        dim3 grid(VV / 128, (TT * BB) / 64);
        gemm_lds_kernel<<<grid, 256, 0, stream>>>(ybf, wywb, wyb, (float*)d_out,
                                                  TT * BB, VV, HH);
    }

    // 6. final hidden state -> output tail [L,B,H]
    hipMemcpyAsync((float*)d_out + (size_t)TT * BB * VV, hstate,
                   (size_t)LL * BB * HH * 4, hipMemcpyDeviceToDevice, stream);
}
